// KerasLSTMLayer_90031104459513
// MI455X (gfx1250) — matrix operand order repaired and checked by reading, not compiled, not run on hardware
//
#include <hip/hip_runtime.h>

// ---- problem constants (match reference) ----
#define UNITS     1024
#define INPUT_DIM 1024
#define BATCH     64
#define TIME      512
#define KTOT      (INPUT_DIM + UNITS)   // 2048 combined K for [x ; h] @ [W ; R]
#define NTOT      (4 * UNITS)           // 4096 gate columns, order i,f,g,o
#define KSPLIT    4                     // waves per block, each reduces K/4 = 512

typedef __attribute__((ext_vector_type(16))) __bf16 v16bf;
typedef __attribute__((ext_vector_type(8)))  float  v8f;

// -------------------- prep kernels --------------------

// fp32 -> bf16 elementwise (inputs)
__global__ void cvt_f32_bf16(const float* __restrict__ src,
                             __bf16* __restrict__ dst, int n) {
  int i = blockIdx.x * blockDim.x + threadIdx.x;
  int stride = gridDim.x * blockDim.x;
  for (; i < n; i += stride) dst[i] = (__bf16)src[i];
}

// Build combined, transposed bf16 weight: Wt[n][k], n in [0,4096), k in [0,2048)
// k <  1024 : kernel[k][n]          (input projection)
// k >= 1024 : recurrent[k-1024][n]  (recurrent projection)
__global__ void build_wt(const float* __restrict__ W,
                         const float* __restrict__ R,
                         __bf16* __restrict__ Wt) {
  int i = blockIdx.x * blockDim.x + threadIdx.x;
  int stride = gridDim.x * blockDim.x;
  const int total = NTOT * KTOT;
  for (; i < total; i += stride) {
    int n = i >> 11;          // / KTOT
    int k = i & (KTOT - 1);   // % KTOT
    float v = (k < INPUT_DIM) ? W[(size_t)k * NTOT + n]
                              : R[(size_t)(k - INPUT_DIM) * NTOT + n];
    Wt[i] = (__bf16)v;
  }
}

// zero initial cell state and h0 (bf16)
__global__ void init_state(float* __restrict__ c_ws, __bf16* __restrict__ h0) {
  int i = blockIdx.x * blockDim.x + threadIdx.x;
  if (i < BATCH * UNITS) { c_ws[i] = 0.0f; h0[i] = (__bf16)0.0f; }
}

// -------------------- fused LSTM step --------------------
// Block = 128 threads = 4 waves, one 16x16 (batch x unit) tile per block.
// Each wave reduces a 512-wide slice of combined K (x-halves for waves 0..1,
// h-halves for waves 2..3), holding 4 gate accumulators. Partials are summed
// through LDS, then a flat elementwise pass does the fp32 LSTM cell math.
// grid = (BATCH/16, UNITS/16) = 256 blocks -> 1024 waves per step.
__global__ __launch_bounds__(128)
void lstm_step(const __bf16* __restrict__ Xt,    // inputs bf16, offset by t*INPUT_DIM; batch stride TIME*INPUT_DIM
               const __bf16* __restrict__ Hin,   // [BATCH][UNITS] bf16
               const __bf16* __restrict__ Wt,    // [NTOT][KTOT] bf16 (transposed combined weights)
               const float*  __restrict__ bias,  // [NTOT] fp32
               float*        __restrict__ c_ws,  // cell state scratch, linear per tile
               __bf16*       __restrict__ Hout,  // [BATCH][UNITS] bf16 (next step)
               float*        __restrict__ Out)   // output, offset by t*UNITS; batch stride TIME*UNITS
{
  const int tid  = threadIdx.x;
  const int wav  = tid >> 5;        // 0..3  (K-split slice)
  const int lane = tid & 31;        // 0..31
  const int m0   = blockIdx.x * 16; // batch tile base
  const int u0   = blockIdx.y * 16; // unit tile base
  const int nlo  = lane & 15;       // column within tile / A row
  const int hi   = lane >> 4;       // K-half selector within a fragment

  // partials: [wave][gate][v*32 + lane]  (consecutive lanes -> consecutive banks)
  __shared__ float part[KSPLIT][4][256];

  v8f acc[4];
#pragma unroll
  for (int g = 0; g < 4; ++g)
#pragma unroll
    for (int v = 0; v < 8; ++v) acc[g][v] = 0.0f;

  const int arow = m0 + nlo;          // A-matrix row = batch index
  const int koff = (wav & 1) * 512;   // offset within x (or h) K-range

  // A slice base: waves 0,1 read x_t; waves 2,3 read h
  const __bf16* abase = (wav < 2)
      ? (Xt  + (size_t)arow * ((size_t)TIME * INPUT_DIM) + koff + hi * 8)
      : (Hin + (size_t)arow * UNITS                      + koff + hi * 8);

  // B row pointers (Wt is [N][K]); this wave's combined-K base
  const int kbase = wav * 512;
  const __bf16* brow[4];
#pragma unroll
  for (int g = 0; g < 4; ++g)
    brow[g] = Wt + (size_t)(g * UNITS + u0 + nlo) * KTOT + kbase + hi * 8;

  union Frag { v16bf v; uint4 q[2]; };

#pragma unroll 4
  for (int kc = 0; kc < 512 / 32; ++kc) {
    const int k0 = kc * 32;
    Frag a;
    a.q[0] = *(const uint4*)(abase + k0);        // K k0..k0+7   (this lane's half)
    a.q[1] = *(const uint4*)(abase + k0 + 16);   // K k0+16..+23 (this lane's half)
#pragma unroll
    for (int g = 0; g < 4; ++g) {
      Frag b;
      b.q[0] = *(const uint4*)(brow[g] + k0);
      b.q[1] = *(const uint4*)(brow[g] + k0 + 16);
      acc[g] = __builtin_amdgcn_wmma_f32_16x16x32_bf16(
          false, a.v, false, b.v, (short)0, acc[g], false, false);
    }
  }

  // dump partial accumulators to LDS
#pragma unroll
  for (int g = 0; g < 4; ++g)
#pragma unroll
    for (int v = 0; v < 8; ++v)
      part[wav][g][v * 32 + lane] = acc[g][v];

  __syncthreads();

  // ---- elementwise LSTM cell: 128 threads x 2 elements of the 16x16 tile ----
  const size_t tile = (size_t)(blockIdx.x * (UNITS / 16) + blockIdx.y) * 256;

#pragma unroll
  for (int r = 0; r < 2; ++r) {
    const int e = tid + r * 128;     // 0..255
    const int m = e >> 4;
    const int n = e & 15;
    // producing location in accumulator layout: lane = n + 16*(m>=8), v = m&7
    const int src = (m & 7) * 32 + n + ((m >> 3) << 4);

    float z[4];
#pragma unroll
    for (int g = 0; g < 4; ++g)
      z[g] = bias[g * UNITS + u0 + n]
           + part[0][g][src] + part[1][g][src]
           + part[2][g][src] + part[3][g][src];

    const float ig = 1.0f / (1.0f + __expf(-z[0]));
    const float fg = 1.0f / (1.0f + __expf(-z[1]));
    const float gg = tanhf(z[2]);
    const float og = 1.0f / (1.0f + __expf(-z[3]));

    const float cn = fg * c_ws[tile + e] + ig * gg;
    c_ws[tile + e] = cn;
    const float hv = og * tanhf(cn);

    const int b = m0 + m;
    const int u = u0 + n;
    Out[(size_t)b * ((size_t)TIME * UNITS) + u] = hv;
    Hout[(size_t)b * UNITS + u] = (__bf16)hv;
  }
}

// -------------------- host launcher --------------------
extern "C" void kernel_launch(void* const* d_in, const int* in_sizes, int n_in,
                              void* d_out, int out_size, void* d_ws, size_t ws_size,
                              hipStream_t stream) {
  const float* inputs  = (const float*)d_in[0]; // [B, T, D]
  const float* kernelW = (const float*)d_in[1]; // [D, 4U]
  const float* rkernel = (const float*)d_in[2]; // [U, 4U]
  const float* bias    = (const float*)d_in[3]; // [4U]
  float* out = (float*)d_out;                   // [B, T, U]

  // workspace carve-out (~85 MB total)
  char* ws = (char*)d_ws;
  __bf16* X16 = (__bf16*)ws; ws += (size_t)BATCH * TIME * INPUT_DIM * sizeof(__bf16); // 64 MB
  __bf16* Wt  = (__bf16*)ws; ws += (size_t)NTOT * KTOT * sizeof(__bf16);              // 16 MB
  float*  cst = (float*)ws;  ws += (size_t)BATCH * UNITS * sizeof(float);             // 256 KB
  __bf16* h0  = (__bf16*)ws; ws += (size_t)BATCH * UNITS * sizeof(__bf16);            // 128 KB
  __bf16* h1  = (__bf16*)ws;                                                          // 128 KB

  const int nX = BATCH * TIME * INPUT_DIM;
  cvt_f32_bf16<<<2048, 256, 0, stream>>>(inputs, X16, nX);
  build_wt<<<2048, 256, 0, stream>>>(kernelW, rkernel, Wt);
  init_state<<<(BATCH * UNITS + 255) / 256, 256, 0, stream>>>(cst, h0);

  __bf16* hb[2] = {h0, h1};
  dim3 grid(BATCH / 16, UNITS / 16); // 256 blocks x 4 waves per step
  for (int t = 0; t < TIME; ++t) {
    lstm_step<<<grid, 128, 0, stream>>>(
        X16 + (size_t)t * INPUT_DIM,
        hb[t & 1], Wt, bias, cst, hb[(t + 1) & 1],
        out + (size_t)t * UNITS);
  }
}
